// DifferentiableMemoryBank_24043226923369
// MI455X (gfx1250) — compile-verified, matrix operand order checked by reading, not run on hardware
//
#include <hip/hip_runtime.h>

typedef __attribute__((ext_vector_type(16))) __bf16        v16bf;
typedef __attribute__((ext_vector_type(8)))  float         v8f;
typedef __attribute__((ext_vector_type(4)))  float         f4;
typedef __attribute__((ext_vector_type(4)))  unsigned int  u4;
typedef __attribute__((ext_vector_type(4)))  unsigned short us4;

#define S_DIM 4096
#define D_DIM 256
#define B_DIM 2048

__device__ __forceinline__ unsigned short f2bf(float f) {
  unsigned int u = __float_as_uint(f);
  u += 0x7FFFu + ((u >> 16) & 1u);           // round-to-nearest-even
  return (unsigned short)(u >> 16);
}
__device__ __forceinline__ float sigm(float x) { return 1.0f / (1.0f + __expf(-x)); }

union FragBF { u4 u[2]; v16bf v; };

// C[M,N] = epilogue( scale * A[M,K] @ op(B) )   op(B) = B[N,K]^T (TRANSB) or B[K,N]
template<bool TRANSB, bool SIG>
__global__ __launch_bounds__(256) void gemm_bf16_wmma(
    const float* __restrict__ A, const float* __restrict__ Bm,
    const float* __restrict__ bias, const float* __restrict__ sub,
    const float* __restrict__ mul, float* __restrict__ gsum,
    float* __restrict__ C, int M, int N, int K, float scale)
{
  constexpr int BM = 128, BN = 128, BK = 32;
  __shared__ __align__(16) unsigned short sA[BM * BK];
  __shared__ __align__(16) unsigned short sB[BN * BK];

  const int tid  = threadIdx.x;
  const int wave = tid >> 5, lane = tid & 31;
  const int half = lane >> 4, lrow = lane & 15;
  const int wm = (wave & 3) * 32;   // 4 waves tile M
  const int wn = (wave >> 2) * 64;  // 2 waves tile N
  const int m0 = blockIdx.y * BM;
  const int n0 = blockIdx.x * BN;

  v8f acc[2][4];
  #pragma unroll
  for (int i = 0; i < 2; ++i)
    #pragma unroll
    for (int j = 0; j < 4; ++j)
      #pragma unroll
      for (int r = 0; r < 8; ++r) acc[i][j][r] = 0.0f;

  for (int kt = 0; kt < K; kt += BK) {
    // ---- stage A tile (BM x BK), f32 -> bf16 into LDS ----
    for (int i = tid; i < (BM * BK) / 4; i += 256) {
      int e = i * 4;
      int r = e >> 5;          // / BK
      int kk = e & 31;
      f4 f = *(const f4*)(A + (size_t)(m0 + r) * K + kt + kk);
      us4 p; p[0] = f2bf(f[0]); p[1] = f2bf(f[1]); p[2] = f2bf(f[2]); p[3] = f2bf(f[3]);
      *(us4*)(&sA[r * BK + kk]) = p;
    }
    // ---- stage B tile as [n][k] ----
    if (TRANSB) {
      for (int i = tid; i < (BN * BK) / 4; i += 256) {
        int e = i * 4;
        int r = e >> 5;
        int kk = e & 31;
        f4 f = *(const f4*)(Bm + (size_t)(n0 + r) * K + kt + kk);
        us4 p; p[0] = f2bf(f[0]); p[1] = f2bf(f[1]); p[2] = f2bf(f[2]); p[3] = f2bf(f[3]);
        *(us4*)(&sB[r * BK + kk]) = p;
      }
    } else {
      for (int i = tid; i < (BK * BN) / 4; i += 256) {
        int e = i * 4;
        int r = e >> 7;          // k row
        int nn = e & 127;
        f4 f = *(const f4*)(Bm + (size_t)(kt + r) * N + n0 + nn);
        sB[(nn + 0) * BK + r] = f2bf(f[0]);
        sB[(nn + 1) * BK + r] = f2bf(f[1]);
        sB[(nn + 2) * BK + r] = f2bf(f[2]);
        sB[(nn + 3) * BK + r] = f2bf(f[3]);
      }
    }
    __syncthreads();

    // ---- fragments per ISA 16-bit A layout: row = lane%16, K halves by lane/16 ----
    FragBF fa[2], fb[4];
    #pragma unroll
    for (int i = 0; i < 2; ++i) {
      int r = wm + i * 16 + lrow;
      fa[i].u[0] = *(const u4*)(&sA[r * BK + half * 8]);
      fa[i].u[1] = *(const u4*)(&sA[r * BK + 16 + half * 8]);
    }
    #pragma unroll
    for (int j = 0; j < 4; ++j) {
      int r = wn + j * 16 + lrow;
      fb[j].u[0] = *(const u4*)(&sB[r * BK + half * 8]);
      fb[j].u[1] = *(const u4*)(&sB[r * BK + 16 + half * 8]);
    }
    #pragma unroll
    for (int i = 0; i < 2; ++i)
      #pragma unroll
      for (int j = 0; j < 4; ++j)
        acc[i][j] = __builtin_amdgcn_wmma_f32_16x16x32_bf16(
            false, fa[i].v, false, fb[j].v, (short)0, acc[i][j], false, false);
    __syncthreads();
  }

  // ---- epilogue: C layout lane%16 = N, VGPR r + 8*(lane/16) = M ----
  float lsum = 0.0f;
  #pragma unroll
  for (int i = 0; i < 2; ++i) {
    #pragma unroll
    for (int j = 0; j < 4; ++j) {
      #pragma unroll
      for (int r = 0; r < 8; ++r) {
        int gm = m0 + wm + i * 16 + half * 8 + r;
        int gn = n0 + wn + j * 16 + lrow;
        float v = acc[i][j][r] * scale;
        if (bias) v += bias[gn];
        if (sub)  v -= 0.1f * sub[gn];
        if (SIG)  v = sigm(v);
        if (gsum) lsum += v;
        if (mul)  v *= mul[(size_t)gm * N + gn];
        C[(size_t)gm * N + gn] = v;
      }
    }
  }
  if (gsum) atomicAdd(gsum, lsum);
}

__global__ __launch_bounds__(256) void softmax_rows(
    const float* __restrict__ in, float* __restrict__ out, int n)
{
  __shared__ float red[256];
  const int row = blockIdx.x, tid = threadIdx.x;
  const float* x = in + (size_t)row * n;
  float m = -3.4e38f;
  for (int i = tid; i < n; i += 256) m = fmaxf(m, x[i]);
  red[tid] = m; __syncthreads();
  for (int s = 128; s > 0; s >>= 1) { if (tid < s) red[tid] = fmaxf(red[tid], red[tid + s]); __syncthreads(); }
  m = red[0]; __syncthreads();
  float sum = 0.0f;
  for (int i = tid; i < n; i += 256) sum += __expf(x[i] - m);
  red[tid] = sum; __syncthreads();
  for (int s = 128; s > 0; s >>= 1) { if (tid < s) red[tid] += red[tid + s]; __syncthreads(); }
  float inv = 1.0f / red[0];
  float* y = out + (size_t)row * n;
  for (int i = tid; i < n; i += 256) y[i] = __expf(x[i] - m) * inv;
}

// lv = ligand * sigmoid(cos(phase - pi) * w_volt + b_volt)
__global__ void nmda_lv(const float* __restrict__ ligand, const float* __restrict__ phase,
                        const float* __restrict__ w_volt, const float* __restrict__ b_volt,
                        float* __restrict__ lv)
{
  int idx = blockIdx.x * blockDim.x + threadIdx.x;
  if (idx >= B_DIM * D_DIM) return;
  int b = idx >> 8, d = idx & 255;
  float ps = __cosf(phase[b] - 3.14159265358979323846f);
  lv[idx] = ligand[idx] * sigm(ps * w_volt[d] + b_volt[d]);
}

// register-resident sequential scan: each thread owns (d fixed, 32 s-values)
__global__ __launch_bounds__(256) void memory_update(
    const float* __restrict__ mem, const float* __restrict__ aw,
    const float* __restrict__ er, const float* __restrict__ ad,
    float* __restrict__ out)
{
  constexpr int CH = 64, TS = 32;
  __shared__ float sw[CH * TS];
  const int tid = threadIdx.x;
  const int d = tid;                       // blockDim == D
  const int s0 = blockIdx.x * TS;
  float m[TS];
  #pragma unroll
  for (int j = 0; j < TS; ++j) m[j] = mem[(size_t)(s0 + j) * D_DIM + d];

  for (int bc = 0; bc < B_DIM; bc += CH) {
    __syncthreads();
    for (int i = tid; i < CH * TS; i += 256)
      sw[i] = aw[(size_t)(bc + (i >> 5)) * S_DIM + s0 + (i & 31)];
    __syncthreads();
    if (bc + CH < B_DIM) {
      __builtin_prefetch(&er[(size_t)(bc + CH) * D_DIM + d], 0, 1);
      __builtin_prefetch(&ad[(size_t)(bc + CH) * D_DIM + d], 0, 1);
    }
    for (int b = 0; b < CH; ++b) {
      float e = er[(size_t)(bc + b) * D_DIM + d];
      float a = ad[(size_t)(bc + b) * D_DIM + d];
      #pragma unroll
      for (int j = 0; j < TS; ++j) {
        float w = sw[b * TS + j];
        m[j] = fmaf(-w * e, m[j], fmaf(w, a, m[j]));   // m*(1-we) + wa
      }
    }
  }
  #pragma unroll
  for (int j = 0; j < TS; ++j) out[(size_t)(s0 + j) * D_DIM + d] = m[j];
}

__global__ void usage_update(const float* __restrict__ usage, const float* __restrict__ aw,
                             float* __restrict__ nu)
{
  int s = blockIdx.x * blockDim.x + threadIdx.x;
  if (s >= S_DIM) return;
  float acc = usage[s];
  for (int b = 0; b < B_DIM; ++b) acc += aw[(size_t)b * S_DIM + s];
  nu[s] = acc;
}

__global__ void zero_one(float* p) { *p = 0.0f; }
__global__ void fin_gate(const float* g, float* out) { *out = *g * (1.0f / (float)(B_DIM * D_DIM)); }

extern "C" void kernel_launch(void* const* d_in, const int* in_sizes, int n_in,
                              void* d_out, int out_size, void* d_ws, size_t ws_size,
                              hipStream_t stream) {
  const float* query   = (const float*)d_in[0];
  const float* content = (const float*)d_in[1];
  const float* phase   = (const float*)d_in[2];
  const float* memory  = (const float*)d_in[3];
  const float* usage   = (const float*)d_in[4];
  const float* w_lig   = (const float*)d_in[5];
  const float* b_lig   = (const float*)d_in[6];
  const float* w_volt  = (const float*)d_in[7];
  const float* b_volt  = (const float*)d_in[8];
  const float* w_gate  = (const float*)d_in[9];
  const float* b_gate  = (const float*)d_in[10];
  const float* w_rq    = (const float*)d_in[11];
  const float* b_rq    = (const float*)d_in[12];
  const float* w_rk    = (const float*)d_in[13];
  const float* b_rk    = (const float*)d_in[14];
  const float* w_rv    = (const float*)d_in[15];
  const float* b_rv    = (const float*)d_in[16];
  const float* w_addr  = (const float*)d_in[17];
  const float* b_addr  = (const float*)d_in[18];
  const float* w_erase = (const float*)d_in[19];
  const float* b_erase = (const float*)d_in[20];
  const float* w_add   = (const float*)d_in[21];
  const float* b_add   = (const float*)d_in[22];

  float* ws  = (float*)d_ws;
  float* out = (float*)d_out;

  // workspace layout (floats)
  const size_t BD = (size_t)B_DIM * D_DIM;     // 524288
  const size_t SD = (size_t)S_DIM * D_DIM;     // 1048576
  const size_t BS = (size_t)B_DIM * S_DIM;     // 8388608
  float* ws_ligand = ws;                 // BD
  float* ws_lv     = ws_ligand + BD;     // BD
  float* ws_gated  = ws_lv     + BD;     // BD
  float* ws_q      = ws_gated  + BD;     // BD
  float* ws_k      = ws_q      + BD;     // SD
  float* ws_v      = ws_k      + SD;     // SD
  float* ws_scores = ws_v      + SD;     // BS
  float* ws_erase  = ws_scores + BS;     // BD
  float* ws_add    = ws_erase  + BD;     // BD
  float* ws_gsum   = ws_add    + BD;     // 1

  // output layout (floats, concatenated in return order)
  float* out_rc = out;                   // (B, D)
  float* out_rw = out_rc + BD;           // (B, S)
  float* out_gv = out_rw + BS;           // scalar
  float* out_aw = out_gv + 1;            // (B, S)
  float* out_nm = out_aw + BS;           // (S, D)
  float* out_nu = out_nm + SD;           // (S,)

  const dim3 blk(256);
  const dim3 gBD(D_DIM / 128, B_DIM / 128);   // (2, 16)  M=B, N=D
  const dim3 gSD(D_DIM / 128, S_DIM / 128);   // (2, 32)  M=S, N=D
  const dim3 gBS(S_DIM / 128, B_DIM / 128);   // (32, 16) M=B, N=S

  zero_one<<<1, 1, 0, stream>>>(ws_gsum);

  // ligand = sigmoid(content @ w_lig^T + b_lig)
  gemm_bf16_wmma<true, true><<<gBD, blk, 0, stream>>>(
      content, w_lig, b_lig, nullptr, nullptr, nullptr, ws_ligand,
      B_DIM, D_DIM, D_DIM, 1.0f);
  // lv = ligand * voltage(phase)
  nmda_lv<<<(B_DIM * D_DIM) / 256, blk, 0, stream>>>(ws_ligand, phase, w_volt, b_volt, ws_lv);
  // gated = content * sigmoid(lv @ w_gate^T + b_gate); accumulate gate mean
  gemm_bf16_wmma<true, true><<<gBD, blk, 0, stream>>>(
      ws_lv, w_gate, b_gate, nullptr, content, ws_gsum, ws_gated,
      B_DIM, D_DIM, D_DIM, 1.0f);

  // q, k, v projections
  gemm_bf16_wmma<true, false><<<gBD, blk, 0, stream>>>(
      query, w_rq, b_rq, nullptr, nullptr, nullptr, ws_q, B_DIM, D_DIM, D_DIM, 1.0f);
  gemm_bf16_wmma<true, false><<<gSD, blk, 0, stream>>>(
      memory, w_rk, b_rk, nullptr, nullptr, nullptr, ws_k, S_DIM, D_DIM, D_DIM, 1.0f);
  gemm_bf16_wmma<true, false><<<gSD, blk, 0, stream>>>(
      memory, w_rv, b_rv, nullptr, nullptr, nullptr, ws_v, S_DIM, D_DIM, D_DIM, 1.0f);

  // scores = (q @ k^T) / sqrt(D); softmax -> read_weights
  gemm_bf16_wmma<true, false><<<gBS, blk, 0, stream>>>(
      ws_q, ws_k, nullptr, nullptr, nullptr, nullptr, ws_scores,
      B_DIM, S_DIM, D_DIM, 0.0625f);
  softmax_rows<<<B_DIM, blk, 0, stream>>>(ws_scores, out_rw, S_DIM);

  // read_content = read_weights @ v   (B is K-major: v[S,D])
  gemm_bf16_wmma<false, false><<<gBD, blk, 0, stream>>>(
      out_rw, ws_v, nullptr, nullptr, nullptr, nullptr, out_rc,
      B_DIM, D_DIM, S_DIM, 1.0f);

  // logits = gated @ w_addr^T + b_addr - 0.1*usage; softmax -> addr_weights
  gemm_bf16_wmma<true, false><<<gBS, blk, 0, stream>>>(
      ws_gated, w_addr, b_addr, usage, nullptr, nullptr, ws_scores,
      B_DIM, S_DIM, D_DIM, 1.0f);
  softmax_rows<<<B_DIM, blk, 0, stream>>>(ws_scores, out_aw, S_DIM);

  // erase / add heads
  gemm_bf16_wmma<true, true><<<gBD, blk, 0, stream>>>(
      ws_gated, w_erase, b_erase, nullptr, nullptr, nullptr, ws_erase,
      B_DIM, D_DIM, D_DIM, 1.0f);
  gemm_bf16_wmma<true, false><<<gBD, blk, 0, stream>>>(
      ws_gated, w_add, b_add, nullptr, nullptr, nullptr, ws_add,
      B_DIM, D_DIM, D_DIM, 1.0f);

  // new_usage, sequential memory scan, gate mean
  usage_update<<<S_DIM / 256, blk, 0, stream>>>(usage, out_aw, out_nu);
  memory_update<<<S_DIM / 32, blk, 0, stream>>>(memory, out_aw, ws_erase, ws_add, out_nm);
  fin_gate<<<1, 1, 0, stream>>>(ws_gsum, out_gv);
}
